// ConvRNNModel_61675730370665
// MI455X (gfx1250) — matrix operand order repaired and checked by reading, not compiled, not run on hardware
//
#include <hip/hip_runtime.h>
#include <hip/hip_bf16.h>

// ---------------------------------------------------------------------------
// Types for CDNA5 WMMA (wave32): v16bf = 16x16x32 bf16 A/B fragment (8 VGPRs),
// v8f = 16x16 f32 C/D fragment (8 VGPRs).
// ---------------------------------------------------------------------------
typedef __attribute__((ext_vector_type(16))) __bf16        v16bf;
typedef __attribute__((ext_vector_type(8)))  float         v8f;
typedef __attribute__((ext_vector_type(8)))  unsigned short ush8;

union FragU { v16bf bf; ush8 us[2]; };

__device__ __forceinline__ unsigned short f2bf(float f) {
    union { float f; unsigned u; } v; v.f = f;
    unsigned r = v.u + 0x7FFFu + ((v.u >> 16) & 1u);   // round-to-nearest-even
    return (unsigned short)(r >> 16);
}
__device__ __forceinline__ float bf2f(unsigned short h) {
    union { unsigned u; float f; } v; v.u = ((unsigned)h) << 16;
    return v.f;
}
__device__ __forceinline__ float sigm(float x) { return 1.f / (1.f + __expf(-x)); }

// ---------------------------------------------------------------------------
// Problem constants
// ---------------------------------------------------------------------------
#define BB   32
#define TT   256
#define NIMG (BB * TT)          // 8192
#define FDIM 832                // 26*32 conv features
#define HDIM 512                // LSTM hidden
#define GDIM 2048               // 4*HDIM

// ---------------------------------------------------------------------------
// Kernel 1: per-image CNN (conv3x3 valid -> maxpool3 -> BN -> ReLU, 4 layers)
// One workgroup (256 thr) per image. Writes 832 bf16 features per image.
// VALU-shaped on purpose: 1->6->6->6->1 channels is not a tensor-core problem.
// ---------------------------------------------------------------------------
__device__ void conv3x3(const float* __restrict__ in, float* __restrict__ out,
                        const float* __restrict__ wt, const float* __restrict__ bias,
                        int Ci, int Co, int Hin, int Win, int tid) {
    int Ho = Hin - 2, Wo = Win - 2, hw = Ho * Wo, tot = Co * hw;
    for (int i = tid; i < tot; i += 256) {
        int co = i / hw, rem = i - co * hw;
        int y = rem / Wo, xx = rem - y * Wo;
        float acc = bias[co];
        for (int ci = 0; ci < Ci; ++ci) {
            const float* ip = in + ci * Hin * Win + y * Win + xx;
            const float* wp = wt + (co * Ci + ci) * 9;
            acc += ip[0] * wp[0] + ip[1] * wp[1] + ip[2] * wp[2]
                 + ip[Win] * wp[3] + ip[Win + 1] * wp[4] + ip[Win + 2] * wp[5]
                 + ip[2 * Win] * wp[6] + ip[2 * Win + 1] * wp[7] + ip[2 * Win + 2] * wp[8];
        }
        out[i] = acc;
    }
}

__device__ void pool_bn_relu(const float* __restrict__ in, float* __restrict__ out,
                             const float* __restrict__ scale, const float* __restrict__ shift,
                             int C, int Hin, int Win, int tid) {
    int Ho = Hin - 2, Wo = Win - 2, hw = Ho * Wo, tot = C * hw;
    for (int i = tid; i < tot; i += 256) {
        int c = i / hw, rem = i - c * hw;
        int y = rem / Wo, xx = rem - y * Wo;
        const float* ip = in + c * Hin * Win + y * Win + xx;
        float mx = ip[0];
        mx = fmaxf(mx, ip[1]);            mx = fmaxf(mx, ip[2]);
        mx = fmaxf(mx, ip[Win]);          mx = fmaxf(mx, ip[Win + 1]);
        mx = fmaxf(mx, ip[Win + 2]);      mx = fmaxf(mx, ip[2 * Win]);
        mx = fmaxf(mx, ip[2 * Win + 1]);  mx = fmaxf(mx, ip[2 * Win + 2]);
        float v = mx * scale[c] + shift[c];
        out[i] = v > 0.f ? v : 0.f;
    }
}

__device__ void stage_layer(const float* __restrict__ w, int nw,
                            const float* __restrict__ b, const float* __restrict__ g,
                            const float* __restrict__ be, const float* __restrict__ m,
                            const float* __restrict__ v, int C,
                            float* __restrict__ wbf, int tid) {
    for (int i = tid; i < nw; i += 256) wbf[i] = w[i];
    if (tid < C) {
        wbf[324 + tid] = b[tid];
        float sc = g[tid] * rsqrtf(v[tid] + 1e-5f);
        wbf[330 + tid] = sc;
        wbf[336 + tid] = be[tid] - m[tid] * sc;
    }
}

__global__ void conv_block_kernel(
    const float* __restrict__ x,
    const float* w1, const float* b1, const float* g1, const float* be1, const float* m1, const float* v1,
    const float* w2, const float* b2, const float* g2, const float* be2, const float* m2, const float* v2,
    const float* w3, const float* b3, const float* g3, const float* be3, const float* m3, const float* v3,
    const float* w4, const float* b4, const float* g4, const float* be4, const float* m4, const float* v4,
    unsigned short* __restrict__ feat) {

    extern __shared__ float smem[];
    float* sin = smem;            // 2016  : 42x48 input
    float* bufA = smem + 2016;    // 11040 : conv outputs (max 6x40x46)
    float* bufB = bufA + 11040;   // 10032 : pooled outputs (max 6x38x44)
    float* wbf  = bufB + 10032;   // 352   : [0..323] w, [324..329] b, [330..335] sc, [336..341] sh

    const int r = blockIdx.x;     // image index in [0, B*T)
    const int tid = threadIdx.x;

    for (int i = tid; i < 42 * 48; i += 256) sin[i] = x[(size_t)r * (42 * 48) + i];
    stage_layer(w1, 54, b1, g1, be1, m1, v1, 6, wbf, tid);
    __syncthreads();
    conv3x3(sin, bufA, wbf, wbf + 324, 1, 6, 42, 48, tid);
    __syncthreads();
    pool_bn_relu(bufA, bufB, wbf + 330, wbf + 336, 6, 40, 46, tid);
    __syncthreads();

    stage_layer(w2, 324, b2, g2, be2, m2, v2, 6, wbf, tid);
    __syncthreads();
    conv3x3(bufB, bufA, wbf, wbf + 324, 6, 6, 38, 44, tid);
    __syncthreads();
    pool_bn_relu(bufA, bufB, wbf + 330, wbf + 336, 6, 36, 42, tid);
    __syncthreads();

    stage_layer(w3, 324, b3, g3, be3, m3, v3, 6, wbf, tid);
    __syncthreads();
    conv3x3(bufB, bufA, wbf, wbf + 324, 6, 6, 34, 40, tid);
    __syncthreads();
    pool_bn_relu(bufA, bufB, wbf + 330, wbf + 336, 6, 32, 38, tid);
    __syncthreads();

    stage_layer(w4, 54, b4, g4, be4, m4, v4, 1, wbf, tid);
    __syncthreads();
    conv3x3(bufB, bufA, wbf, wbf + 324, 6, 1, 30, 36, tid);   // -> 28x34 in bufA
    __syncthreads();

    {   // final maxpool3 + BN + ReLU -> 26x32, store bf16 features
        float sc = wbf[330], sh = wbf[336];
        for (int i = tid; i < 26 * 32; i += 256) {
            int y = i >> 5, xx = i & 31;
            const float* ip = bufA + y * 34 + xx;
            float mx = ip[0];
            mx = fmaxf(mx, ip[1]);   mx = fmaxf(mx, ip[2]);
            mx = fmaxf(mx, ip[34]);  mx = fmaxf(mx, ip[35]);  mx = fmaxf(mx, ip[36]);
            mx = fmaxf(mx, ip[68]);  mx = fmaxf(mx, ip[69]);  mx = fmaxf(mx, ip[70]);
            float v = fmaxf(mx * sc + sh, 0.f);
            feat[(size_t)r * FDIM + i] = f2bf(v);
        }
    }
}

// ---------------------------------------------------------------------------
// Weight conversion / packing kernels
// ---------------------------------------------------------------------------
__global__ void cvt_bf16_kernel(const float* __restrict__ in, unsigned short* __restrict__ out, int n) {
    int i = blockIdx.x * 256 + threadIdx.x;
    if (i < n) out[i] = f2bf(in[i]);
}

// Wcat[n][k], n<2048, k<1024: k<512 -> wih[n][k], else whh[n][k-512]  (bf16)
__global__ void build_wcat_kernel(const float* __restrict__ wih, const float* __restrict__ whh,
                                  unsigned short* __restrict__ wcat) {
    int i = blockIdx.x * 256 + threadIdx.x;
    if (i < GDIM * 1024) {
        int n = i >> 10, k = i & 1023;
        float v = (k < HDIM) ? wih[n * HDIM + k] : whh[n * HDIM + (k - HDIM)];
        wcat[i] = f2bf(v);
    }
}

__global__ void bias_sum_kernel(const float* __restrict__ bih, const float* __restrict__ bhh,
                                float* __restrict__ bsum) {
    int i = blockIdx.x * 256 + threadIdx.x;
    if (i < GDIM) bsum[i] = bih[i] + bhh[i];
}

// ---------------------------------------------------------------------------
// Kernel 2: projection GEMM  proj = feat[8192,832] @ wp^T[832,512] + bp
// bf16 WMMA, f32 accumulate.  One 16x16 tile per wave; 8 waves per WG.
// Output written bf16 into projT laid out [t][b][512] for the LSTM.
// ---------------------------------------------------------------------------
__global__ void proj_gemm_kernel(const unsigned short* __restrict__ feat,
                                 const unsigned short* __restrict__ wpb,   // bf16 wp [512][832]
                                 const float* __restrict__ bp,
                                 unsigned short* __restrict__ projT) {
    const int wv   = threadIdx.x >> 5;
    const int lane = threadIdx.x & 31;
    const int half = lane >> 4;
    const int mr   = lane & 15;

    const int tile  = blockIdx.x * 8 + wv;   // 512 mtiles * 32 ntiles = 16384
    const int mtile = tile >> 5;
    const int ntile = tile & 31;

    const unsigned short* pa = feat + (size_t)(mtile * 16 + mr) * FDIM;
    const unsigned short* pb = wpb  + (size_t)(ntile * 16 + mr) * FDIM;

    v8f acc = {};
    for (int kk = 0; kk < FDIM / 32; ++kk) {
        const int k0 = kk * 32;
        __builtin_prefetch(pa + k0 + 64, 0, 1);
        __builtin_prefetch(pb + k0 + 64, 0, 1);
        FragU a, b;
        // A 16x32 bf16: chunks at K = half*8 and 16+half*8 (ISA layout)
        a.us[0] = *(const ush8*)(pa + k0 + half * 8);
        a.us[1] = *(const ush8*)(pa + k0 + 16 + half * 8);
        // B 32x16 bf16: chunks at K = half*8 and 16+half*8 (same as A)
        b.us[0] = *(const ush8*)(pb + k0 + half * 8);
        b.us[1] = *(const ush8*)(pb + k0 + 16 + half * 8);
        acc = __builtin_amdgcn_wmma_f32_16x16x32_bf16(false, a.bf, false, b.bf,
                                                      (short)0, acc, false, false);
    }

    const int n = ntile * 16 + mr;           // output column (hidden dim)
    const float bias = bp[n];
    const int rowbase = mtile * 16 + half * 8;
    for (int r = 0; r < 8; ++r) {
        int row = rowbase + r;               // flat image index = b*T + t
        int bidx = row >> 8, t = row & 255;
        projT[((size_t)((t << 5) + bidx)) * HDIM + n] = f2bf(acc[r] + bias);
    }
}

// ---------------------------------------------------------------------------
// Kernel 3: LSTM recurrence, persistent single workgroup (1024 thr = 32 waves).
// Per step: gates = [x_t | h] (32x1024 bf16) @ Wcat^T (2048x1024 bf16), WMMA.
// Each wave owns 2 "gate quads": the same 16x16 (b,j) tile in the i/f/g/o
// blocks -> pointwise LSTM update is lane-local, c lives in registers.
// h is shared via 32KB bf16 LDS; two barriers per step.
// ---------------------------------------------------------------------------
__global__ void __launch_bounds__(1024)
lstm_kernel(const unsigned short* __restrict__ projT,   // [T][B][512] bf16
            const unsigned short* __restrict__ wcat,    // [2048][1024] bf16
            const float* __restrict__ bsum,             // [2048] bih+bhh
            const float* __restrict__ wc,               // [2][512]
            const float* __restrict__ bc,               // [2]
            float* __restrict__ out) {                  // [32][2]
    __shared__ unsigned short hbf[BB * HDIM];           // 32 KB

    const int tid  = threadIdx.x;
    const int wv   = tid >> 5;
    const int lane = tid & 31;
    const int half = lane >> 4;
    const int mr   = lane & 15;

    const int mtile = wv >> 4;                // 0..1 -> batch rows 0-15 / 16-31
    const int ntq0  = (wv & 15) * 2;          // 2 quads per wave: ntq0, ntq0+1
    const int arow  = mtile * 16 + mr;        // A-matrix row = batch index

    // Per-lane constant B-row pointers and gate biases (j fixed per lane).
    const unsigned short* wrow[2][4];
    float biasr[2][4];
    for (int q = 0; q < 2; ++q)
        for (int g = 0; g < 4; ++g) {
            int n = g * HDIM + (ntq0 + q) * 16 + mr;
            wrow[q][g]  = wcat + (size_t)n * 1024;
            biasr[q][g] = bsum[n];
        }

    float creg[2][8];
    for (int q = 0; q < 2; ++q)
        for (int r = 0; r < 8; ++r) creg[q][r] = 0.f;
    for (int i = tid; i < BB * HDIM; i += 1024) hbf[i] = 0;
    __syncthreads();

    for (int t = 0; t < TT; ++t) {
        v8f acc[2][4];
        const v8f z = {};
        for (int q = 0; q < 2; ++q)
            for (int g = 0; g < 4; ++g) acc[q][g] = z;

        const unsigned short* xrow = projT + ((size_t)(t << 5) + arow) * HDIM;

        for (int kk = 0; kk < 32; ++kk) {     // K = 1024 (512 x_t + 512 h)
            FragU a;
            if (kk < 16) {
                const unsigned short* pa = xrow + kk * 32;
                a.us[0] = *(const ush8*)(pa + half * 8);
                a.us[1] = *(const ush8*)(pa + 16 + half * 8);
            } else {
                const unsigned short* pa = hbf + arow * HDIM + (kk - 16) * 32;
                a.us[0] = *(const ush8*)(pa + half * 8);
                a.us[1] = *(const ush8*)(pa + 16 + half * 8);
            }
            for (int q = 0; q < 2; ++q)
                for (int g = 0; g < 4; ++g) {
                    FragU b;
                    const unsigned short* pb = wrow[q][g] + kk * 32;
                    b.us[0] = *(const ush8*)(pb + half * 8);
                    b.us[1] = *(const ush8*)(pb + 16 + half * 8);
                    acc[q][g] = __builtin_amdgcn_wmma_f32_16x16x32_bf16(
                        false, a.bf, false, b.bf, (short)0, acc[q][g], false, false);
                }
        }

        __syncthreads();   // all h reads done before overwriting h

        // Lane-local LSTM pointwise update (i,f,g,o share lane/VGPR slots).
        for (int q = 0; q < 2; ++q) {
            const int j = (ntq0 + q) * 16 + mr;
            for (int r = 0; r < 8; ++r) {
                float gi = acc[q][0][r] + biasr[q][0];
                float gf = acc[q][1][r] + biasr[q][1];
                float gg = acc[q][2][r] + biasr[q][2];
                float go = acc[q][3][r] + biasr[q][3];
                float c  = sigm(gf) * creg[q][r] + sigm(gi) * tanhf(gg);
                creg[q][r] = c;
                float h = sigm(go) * tanhf(c);
                int b = mtile * 16 + half * 8 + r;
                hbf[b * HDIM + j] = f2bf(h);
            }
        }
        __syncthreads();   // new h visible before next step's reads
    }

    // logits = h_T @ wc^T + bc   (32x2)
    if (tid < 64) {
        int b = tid >> 1, cls = tid & 1;
        float s = bc[cls];
        for (int k = 0; k < HDIM; ++k)
            s += bf2f(hbf[b * HDIM + k]) * wc[cls * HDIM + k];
        out[b * 2 + cls] = s;
    }
}

// ---------------------------------------------------------------------------
// Host: workspace layout + launch sequence
// ---------------------------------------------------------------------------
extern "C" void kernel_launch(void* const* d_in, const int* in_sizes, int n_in,
                              void* d_out, int out_size, void* d_ws, size_t ws_size,
                              hipStream_t stream) {
    (void)in_sizes; (void)n_in; (void)out_size; (void)ws_size;

    const float* x   = (const float*)d_in[0];
    const float* w1  = (const float*)d_in[1];
    const float* b1  = (const float*)d_in[2];
    const float* g1  = (const float*)d_in[3];
    const float* be1 = (const float*)d_in[4];
    const float* m1  = (const float*)d_in[5];
    const float* v1  = (const float*)d_in[6];
    const float* w2  = (const float*)d_in[7];
    const float* b2  = (const float*)d_in[8];
    const float* g2  = (const float*)d_in[9];
    const float* be2 = (const float*)d_in[10];
    const float* m2  = (const float*)d_in[11];
    const float* v2  = (const float*)d_in[12];
    const float* w3  = (const float*)d_in[13];
    const float* b3  = (const float*)d_in[14];
    const float* g3  = (const float*)d_in[15];
    const float* be3 = (const float*)d_in[16];
    const float* m3  = (const float*)d_in[17];
    const float* v3  = (const float*)d_in[18];
    const float* w4  = (const float*)d_in[19];
    const float* b4  = (const float*)d_in[20];
    const float* g4  = (const float*)d_in[21];
    const float* be4 = (const float*)d_in[22];
    const float* m4  = (const float*)d_in[23];
    const float* v4  = (const float*)d_in[24];
    const float* wp  = (const float*)d_in[25];
    const float* bp  = (const float*)d_in[26];
    const float* wih = (const float*)d_in[27];
    const float* whh = (const float*)d_in[28];
    const float* bih = (const float*)d_in[29];
    const float* bhh = (const float*)d_in[30];
    const float* wc  = (const float*)d_in[31];
    const float* bc  = (const float*)d_in[32];
    float* out = (float*)d_out;

    char* ws = (char*)d_ws;
    unsigned short* feat  = (unsigned short*)(ws);               // 8192*832*2  = 13,631,488
    unsigned short* projT = (unsigned short*)(ws + 13631488);    // 8192*512*2  =  8,388,608
    unsigned short* wpb   = (unsigned short*)(ws + 22020096);    //  512*832*2  =    851,968
    unsigned short* wcat  = (unsigned short*)(ws + 22872064);    // 2048*1024*2 =  4,194,304
    float*          bsum  = (float*)(ws + 27066368);             // 2048*4      =      8,192

    // Weight packing (bf16)
    cvt_bf16_kernel<<<(512 * 832 + 255) / 256, 256, 0, stream>>>(wp, wpb, 512 * 832);
    build_wcat_kernel<<<(GDIM * 1024 + 255) / 256, 256, 0, stream>>>(wih, whh, wcat);
    bias_sum_kernel<<<(GDIM + 255) / 256, 256, 0, stream>>>(bih, bhh, bsum);

    // CNN feature extraction: one WG per image, dynamic LDS = 23440 floats
    const size_t conv_lds = (2016 + 11040 + 10032 + 352) * sizeof(float);
    conv_block_kernel<<<NIMG, 256, conv_lds, stream>>>(
        x, w1, b1, g1, be1, m1, v1, w2, b2, g2, be2, m2, v2,
        w3, b3, g3, be3, m3, v3, w4, b4, g4, be4, m4, v4, feat);

    // Projection GEMM (WMMA): 16384 tiles / 8 waves per WG
    proj_gemm_kernel<<<2048, 256, 0, stream>>>(feat, wpb, bp, projT);

    // Persistent-WG LSTM recurrence (WMMA) + classifier
    lstm_kernel<<<1, 1024, 0, stream>>>(projT, wcat, bsum, wc, bc, out);
}
